// EnsembleMember_43456479101235
// MI455X (gfx1250) — compile-verified, matrix operand order checked by reading, not run on hardware
//
#include <hip/hip_runtime.h>
#include <hip/hip_bf16.h>

#define B_SZ 256
#define T_SZ 2048
#define DIN  3
#define H    128
#define G4H  512          // 4*H
#define DOUT 26
#define K0   160          // layer0 fused K: 128 (h) + 3 (x) padded to 160
#define K1   256          // layer1 fused K: 128 (h2 state) + 128 (h1_t input)

typedef __attribute__((ext_vector_type(16))) __bf16 v16bf;
typedef __attribute__((ext_vector_type(8)))  __bf16 v8bf;
typedef __attribute__((ext_vector_type(8)))  float  v8f;

struct V8Pair { v8bf lo; v8bf hi; };
union  FragU  { v16bf v; V8Pair p; };

// Load one wave32 WMMA bf16 operand fragment.
// CDNA5 16-bit A/B layout: lane<16 holds K = {off..off+7, off+16..off+23},
// lane>=16 holds K = {off+8..off+15, off+24..off+31}; row-major storage makes
// these two contiguous 16B runs -> two b128 loads.
__device__ __forceinline__ v16bf ldfrag(const __bf16* p) {
    FragU u;
    u.p.lo = *reinterpret_cast<const v8bf*>(p);
    u.p.hi = *reinterpret_cast<const v8bf*>(p + 16);
    return u.v;
}

// sLSTM exponential gating with log-domain stabilizer, elementwise on the
// 8 accumulator lanes of a C/D tile (lane owns col=lane&15, rows e+8*(lane>>4)).
// Gate preactivations arrive as two partial accumulator chains (pa + pb).
__device__ __forceinline__ void gate_update(v8f& c, v8f& n, v8f& m,
                                            const v8f& iA, const v8f& iB,
                                            const v8f& fA, const v8f& fB,
                                            const v8f& zA, const v8f& zB,
                                            const v8f& oA, const v8f& oB,
                                            const float* bc, v8f& hout) {
#pragma unroll
    for (int e = 0; e < 8; ++e) {
        float ip = (iA[e] + iB[e]) + bc[0];
        float fp = (fA[e] + fB[e]) + bc[1];
        float zp = (zA[e] + zB[e]) + bc[2];
        float op = (oA[e] + oB[e]) + bc[3];
        float mn = fmaxf(fp + m[e], ip);
        float ig = __expf(ip - mn);
        float fg = __expf(fp + m[e] - mn);
        float zg = 1.f - 2.f / (__expf(2.f * zp) + 1.f);   // tanh
        float og = 1.f / (1.f + __expf(-op));              // sigmoid
        float cn = fg * c[e] + ig * zg;
        float nn = fg * n[e] + ig;
        c[e] = cn; n[e] = nn; m[e] = mn;
        hout[e] = og * (cn / nn);
    }
}

// One-time weight repack to bf16, fragment-friendly row-major layout.
// ws (bf16 elems): [0, 512*160)           B0 = [R0 | W0 | zero-pad] per gate row
//                  [512*160, +512*128)    R1
//                  [.. , +512*128)        W1
__global__ void slstm_pack_weights(const float* __restrict__ W0,
                                   const float* __restrict__ R0,
                                   const float* __restrict__ W1,
                                   const float* __restrict__ R1,
                                   __bf16* __restrict__ ws) {
    int idx = blockIdx.x * 256 + threadIdx.x;
    if (idx < G4H * K0) {
        int nrow = idx / K0, k = idx % K0;
        float v = (k < H) ? R0[nrow * H + k]
                          : ((k < H + DIN) ? W0[nrow * DIN + (k - H)] : 0.f);
        ws[idx] = (__bf16)v;
    } else if (idx < G4H * K0 + G4H * H) {
        ws[idx] = (__bf16)R1[idx - G4H * K0];
    } else if (idx < G4H * K0 + 2 * G4H * H) {
        ws[idx] = (__bf16)W1[idx - (G4H * K0 + G4H * H)];
    }
}

// Persistent fused scan: 16 blocks (one per 16-row batch tile), 8 waves each.
// Wave w owns hidden units [16w,16w+16) for BOTH layers (gate tiles at columns
// 16w + g*128), so i/f/z/o for a hidden unit stay in one wave and the c/n/m
// state plus all layer-1 B fragments are register-resident.
__global__ __launch_bounds__(256, 1)
void slstm_fused(const float* __restrict__ x,
                 const float* __restrict__ b0,
                 const float* __restrict__ b1,
                 const float* __restrict__ Wmu,  const float* __restrict__ bmu,
                 const float* __restrict__ Wsig, const float* __restrict__ bsig,
                 const __bf16* __restrict__ ws,
                 float* __restrict__ out) {
    extern __shared__ __bf16 smem[];
    __bf16* sB0 = smem;                 // 512*160 bf16 = 160 KB
    __bf16* sA0 = sB0 + G4H * K0;       // 2 x 16*160  (ping-pong [h1 | x_t | 0])
    __bf16* sA1 = sA0 + 2 * 16 * K0;    // 2 x 16*256  (ping-pong [h2 | h1_t])

    const int tid  = threadIdx.x;
    const int wg   = blockIdx.x;        // batch tile
    const int wv   = tid >> 5;          // wave id 0..7 -> hidden block
    const int lane = tid & 31;
    const int hi   = lane >> 4;
    const int col  = lane & 15;
    const int koff = hi * 8;
    const int arow = col;               // A-matrix row for this lane

    // ---- stage B0 global -> LDS (one-time, 16B vector copies)
    {
        const uint4* src = (const uint4*)ws;
        uint4*       dst = (uint4*)smem;
        const int n16 = G4H * K0 * (int)sizeof(__bf16) / 16;
        for (int i = tid; i < n16; i += 256) dst[i] = src[i];
    }
    // zero both ping-pong A tiles (initial h = 0; pad columns stay 0 forever)
    for (int i = tid; i < 2 * 16 * K0 + 2 * 16 * K1; i += 256)
        sA0[i] = (__bf16)0.f;
    __syncthreads();
    // stage x_{t=0} into A0 buffer 0
    if (tid < 48) {
        int r = tid / 3, d = tid % 3;
        sA0[r * K0 + H + d] =
            (__bf16)x[((size_t)(wg * 16 + r) * T_SZ + 0) * DIN + d];
    }

    // per-lane gate columns, biases, B0 base pointers
    int ng[4]; float bc0[4], bc1[4];
    const __bf16* b0p[4];
#pragma unroll
    for (int g = 0; g < 4; ++g) {
        ng[g]  = wv * 16 + col + g * H;
        bc0[g] = b0[ng[g]];
        bc1[g] = b1[ng[g]];
        b0p[g] = sB0 + ng[g] * K0 + koff;
    }

    // preload ALL layer-1 B fragments (R1 + W1) into VGPRs: 32 frags x 8 VGPRs.
    // Layer 1's steady state then touches only A-tile LDS loads + registers.
    // b1f[g][kc] is the B fragment for K-chunk kc of the fused K=256 matmul:
    // kc 0..3 -> R1, kc 4..7 -> W1.
    const __bf16* wsR1 = ws + G4H * K0;
    const __bf16* wsW1 = wsR1 + G4H * H;
    v16bf b1f[4][8];
#pragma unroll
    for (int g = 0; g < 4; ++g)
#pragma unroll
        for (int kc = 0; kc < 4; ++kc) {
            b1f[g][kc]     = ldfrag(&wsR1[ng[g] * H + kc * 32 + koff]);
            b1f[g][kc + 4] = ldfrag(&wsW1[ng[g] * H + kc * 32 + koff]);
        }

    // recurrent state, register-resident (16 rows x 16 hidden units per wave)
    v8f c0 = {}, n0 = {}, m0 = {}, c1 = {}, n1 = {}, m1 = {};

    __syncthreads();

    const int xr = tid / 3, xd = tid % 3;

#pragma unroll 1
    for (int t = 0; t < T_SZ; ++t) {
        const __bf16* a0r = sA0 + (t & 1) * (16 * K0);
        __bf16*       a0w = sA0 + ((t + 1) & 1) * (16 * K0);
        __bf16*       a1r = sA1 + (t & 1) * (16 * K1);
        __bf16*       a1w = sA1 + ((t + 1) & 1) * (16 * K1);

        // issue next-step x load early; latency hides under the layer-0 matmul
        const bool do_x = (tid < 48) && (t + 1 < T_SZ);
        float xval = 0.f;
        if (do_x)
            xval = x[((size_t)(wg * 16 + xr) * T_SZ + (t + 1)) * DIN + xd];

        // ---------- layer 0: [h1 | x_t] @ [R0; W0]^T, K = 160 ----------
        // All 5 A-fragments + streamed B0 fragments; per-gate accumulation is
        // split into two independent chains (even/odd kc) so the serial WMMA
        // dependency depth is 3 instead of 5.
        v16bf a0f[5];
#pragma unroll
        for (int kc = 0; kc < 5; ++kc)
            a0f[kc] = ldfrag(&a0r[arow * K0 + kc * 32 + koff]);
        v8f acA0[4] = {}, acB0[4] = {};
#pragma unroll
        for (int kc = 0; kc < 5; ++kc) {
#pragma unroll
            for (int g = 0; g < 4; ++g) {
                v16bf bfr = ldfrag(b0p[g] + kc * 32);
                if (kc & 1)
                    acB0[g] = __builtin_amdgcn_wmma_f32_16x16x32_bf16(
                        false, a0f[kc], false, bfr, (short)0, acB0[g], false, false);
                else
                    acA0[g] = __builtin_amdgcn_wmma_f32_16x16x32_bf16(
                        false, a0f[kc], false, bfr, (short)0, acA0[g], false, false);
            }
        }
        v8f h1v;
        gate_update(c0, n0, m0,
                    acA0[0], acB0[0], acA0[1], acB0[1],
                    acA0[2], acB0[2], acA0[3], acB0[3], bc0, h1v);

        // commit h1_t: next step's layer-0 state (other buffer) AND this
        // step's layer-1 input (current A1 buffer, consumed after the barrier)
#pragma unroll
        for (int e = 0; e < 8; ++e) {
            __bf16 hb = (__bf16)h1v[e];
            int mrow = e + 8 * hi;
            a0w[mrow * K0 + wv * 16 + col]     = hb;
            a1r[mrow * K1 + H + wv * 16 + col] = hb;
        }
        if (do_x) a0w[xr * K0 + H + xd] = (__bf16)xval;

        __syncthreads();   // sole barrier: h1_t visible to all waves

        // ---------- layer 1: [h2 | h1_t] @ [R1; W1]^T, K = 256 ----------
        // B entirely in registers; 16 ds_load_b128 for A, then 32 WMMAs in
        // 8 independent chains of depth 4 (even/odd kc split per gate).
        v16bf a1f[8];
#pragma unroll
        for (int kc = 0; kc < 8; ++kc)
            a1f[kc] = ldfrag(&a1r[arow * K1 + kc * 32 + koff]);
        v8f acA1[4] = {}, acB1[4] = {};
#pragma unroll
        for (int kc = 0; kc < 8; ++kc) {
#pragma unroll
            for (int g = 0; g < 4; ++g) {
                if (kc & 1)
                    acB1[g] = __builtin_amdgcn_wmma_f32_16x16x32_bf16(
                        false, a1f[kc], false, b1f[g][kc], (short)0, acB1[g], false, false);
                else
                    acA1[g] = __builtin_amdgcn_wmma_f32_16x16x32_bf16(
                        false, a1f[kc], false, b1f[g][kc], (short)0, acA1[g], false, false);
            }
        }
        v8f h2v;
        gate_update(c1, n1, m1,
                    acA1[0], acB1[0], acA1[1], acB1[1],
                    acA1[2], acB1[2], acA1[3], acB1[3], bc1, h2v);

        // h2_t -> other A1 buffer; next iteration's barrier publishes it
#pragma unroll
        for (int e = 0; e < 8; ++e)
            a1w[(e + 8 * hi) * K1 + wv * 16 + col] = (__bf16)h2v[e];
    }
    __syncthreads();

    // ---------- head: mu / sigma from h2_{T-1} (A1 buffer T%2, cols 0..127)
    const __bf16* hfin = sA1 + (T_SZ & 1) * (16 * K1);
    int row = tid >> 4;
    int j   = tid & 15;
#pragma unroll
    for (int rep = 0; rep < 2; ++rep) {
        int dd = j + rep * 16;
        if (dd < DOUT) {
            float smu = bmu[dd], ssg = bsig[dd];
            for (int k = 0; k < H; ++k) {
                float hv = (float)hfin[row * K1 + k];
                smu += hv * Wmu[dd * H + k];
                ssg += hv * Wsig[dd * H + k];
            }
            int gr = wg * 16 + row;
            out[gr * DOUT + dd] = smu;
            float sp = (ssg > 20.f) ? ssg : __logf(1.f + __expf(ssg));
            out[B_SZ * DOUT + gr * DOUT + dd] = sp + 1e-6f;
        }
    }
}

extern "C" void kernel_launch(void* const* d_in, const int* in_sizes, int n_in,
                              void* d_out, int out_size, void* d_ws, size_t ws_size,
                              hipStream_t stream) {
    (void)in_sizes; (void)n_in; (void)out_size; (void)ws_size;
    const float* x   = (const float*)d_in[0];
    const float* W0  = (const float*)d_in[1];
    const float* R0  = (const float*)d_in[2];
    const float* b0  = (const float*)d_in[3];
    const float* W1  = (const float*)d_in[4];
    const float* R1  = (const float*)d_in[5];
    const float* b1  = (const float*)d_in[6];
    const float* Wmu = (const float*)d_in[7];
    const float* bmu = (const float*)d_in[8];
    const float* Wsg = (const float*)d_in[9];
    const float* bsg = (const float*)d_in[10];
    __bf16* ws = (__bf16*)d_ws;

    int totalw = G4H * K0 + 2 * G4H * H;           // 212992 bf16 = 416 KB scratch
    slstm_pack_weights<<<(totalw + 255) / 256, 256, 0, stream>>>(W0, R0, W1, R1, ws);

    size_t shbytes = (size_t)(G4H * K0 + 2 * 16 * K0 + 2 * 16 * K1) * sizeof(__bf16);
    hipFuncSetAttribute((const void*)slstm_fused,
                        hipFuncAttributeMaxDynamicSharedMemorySize, (int)shbytes);
    slstm_fused<<<16, 256, shbytes, stream>>>(x, b0, b1, Wmu, bmu, Wsg, bsg, ws,
                                              (float*)d_out);
}
